// MoEClassifier_56556129354193
// MI455X (gfx1250) — compile-verified, matrix operand order checked by reading, not run on hardware
//
#include <hip/hip_runtime.h>

// ---------------- problem dims ----------------
#define NTOK 16384
#define DIN  256
#define DM   1024
#define DH   2048
#define NE   8
#define NSLOT (2 * NTOK)

// ---------------- GEMM tiling -----------------
#define BM 128
#define BN 128
#define KT 32
#define ROWP 40   // LDS row stride in bf16 elems: 32 + 8 pad -> 80B rows (16B aligned, conflict-free)

typedef __attribute__((ext_vector_type(16))) __bf16 v16bf;
typedef __attribute__((ext_vector_type(8)))  __bf16 v8bf;
typedef __attribute__((ext_vector_type(8)))  float  v8f;
typedef __attribute__((ext_vector_type(4)))  int    v4i;

#if __has_builtin(__builtin_amdgcn_global_load_async_to_lds_b128) && \
    __has_builtin(__builtin_amdgcn_s_wait_asynccnt)
#define ASYNC_COPY 1
#endif

__device__ __forceinline__ void cp_b128(const void* g, void* l) {
#ifdef ASYNC_COPY
  __builtin_amdgcn_global_load_async_to_lds_b128(
      (__attribute__((address_space(1))) v4i*)g,
      (__attribute__((address_space(3))) v4i*)l, 0, 0);
#else
  *(uint4*)l = *(const uint4*)g;
#endif
}

__device__ __forceinline__ void stage_wait() {
#ifdef ASYNC_COPY
  __builtin_amdgcn_s_wait_asynccnt(0);
#endif
  __syncthreads();
}

// One 16x32 bf16 fragment from an LDS tile stored row-major with K contiguous.
// A-layout (ISA 7.12.2): lane L holds row (L&15); K elems 0..7 -> K = (L>>4)*8 + i,
// elems 8..15 -> K = 16 + (L>>4)*8 + (i-8). B tiles are stored N-major so the
// same loader produces the (symmetric) B fragment.
__device__ __forceinline__ v16bf frag_load(const __bf16* lds, int rowbase, int lane) {
  const int r = rowbase + (lane & 15);
  const int koff = (lane >> 4) * 8;
  const __bf16* p = lds + r * ROWP;
  union { v16bf v; v8bf h[2]; } u;
  u.h[0] = *(const v8bf*)(p + koff);
  u.h[1] = *(const v8bf*)(p + 16 + koff);
  return u.v;
}

__device__ __forceinline__ float gelu_tanh(float x) {
  return 0.5f * x * (1.0f + tanhf(0.7978845608028654f * (x + 0.044715f * x * x * x)));
}

__device__ __forceinline__ float wave_sum32(float v, int lane) {
#pragma unroll
  for (int off = 16; off > 0; off >>= 1)
    v += __int_as_float(__builtin_amdgcn_ds_bpermute((lane ^ off) << 2, __float_as_int(v)));
  return v;
}

__device__ __forceinline__ void block_reduce2(float a, float b, float* r1, float* r2,
                                              float& oa, float& ob, int t) {
  r1[t] = a; r2[t] = b;
  __syncthreads();
  for (int o = 128; o > 0; o >>= 1) {
    if (t < o) { r1[t] += r1[t + o]; r2[t] += r2[t + o]; }
    __syncthreads();
  }
  oa = r1[0]; ob = r2[0];
  __syncthreads();
}

// ---------------- small utility kernels ----------------
__global__ void zero_f32_kernel(float4* p, int n4) {
  int i = blockIdx.x * blockDim.x + threadIdx.x;
  const int stride = gridDim.x * blockDim.x;
  for (; i < n4; i += stride) p[i] = make_float4(0.f, 0.f, 0.f, 0.f);
}

__global__ void f32_to_bf16_kernel(const float* __restrict__ s, __bf16* __restrict__ d, int n) {
  const int i = blockIdx.x * 256 + threadIdx.x;
  if (i < n) d[i] = (__bf16)s[i];
}

// dst[b][c][r] = src[b][r][c]  (weights -> N-major, K-contiguous bf16)
__global__ void transpose_to_bf16_kernel(const float* __restrict__ src, __bf16* __restrict__ dst,
                                         int R, int C) {
  const float* s = src + (size_t)blockIdx.z * R * C;
  __bf16* d = dst + (size_t)blockIdx.z * R * C;
  const int r = blockIdx.x * 256 + threadIdx.x;
  const int c = blockIdx.y;
  if (r < R) d[(size_t)c * R + r] = (__bf16)s[(size_t)r * C + c];
}

// ---------------- input projection GEMM: hpre = x @ w_in + b_in ----------------
__global__ __launch_bounds__(256) void gemm_in_kernel(
    const __bf16* __restrict__ A,   // [NTOK][DIN]
    const __bf16* __restrict__ Bt,  // [DM][DIN]  (N-major)
    const float* __restrict__ bias, // [DM]
    float* __restrict__ C) {        // [NTOK][DM]
  __shared__ __bf16 As[BM * ROWP];
  __shared__ __bf16 Bs[BN * ROWP];
  const int tid = threadIdx.x, lane = tid & 31, wid = tid >> 5;
  const int wm = wid >> 2, wn = wid & 3;
  const int mtile = blockIdx.x * BM, ntile = blockIdx.y * BN;
  v8f acc[4][2] = {};
  for (int kb = 0; kb < DIN; kb += KT) {
#pragma unroll
    for (int it = 0; it < 2; ++it) {
      const int ch = tid + it * 256, row = ch >> 2, cc = ch & 3;
      cp_b128(A + (size_t)(mtile + row) * DIN + kb + cc * 8, As + row * ROWP + cc * 8);
      cp_b128(Bt + (size_t)(ntile + row) * DIN + kb + cc * 8, Bs + row * ROWP + cc * 8);
    }
    stage_wait();
    v16bf af[4], bfr[2];
#pragma unroll
    for (int i = 0; i < 4; ++i) af[i] = frag_load(As, wm * 64 + i * 16, lane);
#pragma unroll
    for (int j = 0; j < 2; ++j) bfr[j] = frag_load(Bs, wn * 32 + j * 16, lane);
#pragma unroll
    for (int i = 0; i < 4; ++i)
#pragma unroll
      for (int j = 0; j < 2; ++j)
        acc[i][j] = __builtin_amdgcn_wmma_f32_16x16x32_bf16(false, af[i], false, bfr[j],
                                                            (short)0, acc[i][j], false, false);
    __syncthreads();
  }
#pragma unroll
  for (int i = 0; i < 4; ++i)
#pragma unroll
    for (int j = 0; j < 2; ++j) {
      const int mbase = mtile + wm * 64 + i * 16 + ((lane >> 4) << 3);
      const int col = ntile + wn * 32 + j * 16 + (lane & 15);
      const float b = bias[col];
#pragma unroll
      for (int r = 0; r < 8; ++r)
        C[(size_t)(mbase + r) * DM + col] = acc[i][j][r] + b;
    }
}

// ---------------- layernorm of input projection ----------------
__global__ __launch_bounds__(256) void ln_in_kernel(
    const float* __restrict__ hpre, const float* __restrict__ g, const float* __restrict__ be,
    float* __restrict__ h, __bf16* __restrict__ hb) {
  const int n = blockIdx.x, t = threadIdx.x;
  __shared__ float r1[256], r2[256];
  float v[4];
  float s = 0.f, ss = 0.f;
#pragma unroll
  for (int j = 0; j < 4; ++j) {
    const float x = hpre[(size_t)n * DM + t + 256 * j];
    v[j] = x; s += x; ss += x * x;
  }
  float S, SS;
  block_reduce2(s, ss, r1, r2, S, SS, t);
  const float mean = S * (1.f / DM);
  const float rs = rsqrtf(SS * (1.f / DM) - mean * mean + 1e-5f);
#pragma unroll
  for (int j = 0; j < 4; ++j) {
    const int d = t + 256 * j;
    const float y = (v[j] - mean) * rs * g[d] + be[d];
    h[(size_t)n * DM + d] = y;
    hb[(size_t)n * DM + d] = (__bf16)y;
  }
}

// ---------------- router: softmax(h @ w_gate) -> top-2, renormalized ----------------
__global__ __launch_bounds__(256) void router_kernel(
    const float* __restrict__ h, const float* __restrict__ w_gate,
    int2* __restrict__ tidx, float2* __restrict__ tgate) {
  const int lane = threadIdx.x & 31;
  const int wave = threadIdx.x >> 5;
  const int n = blockIdx.x * 8 + wave;
  const float* row = h + (size_t)n * DM;
  float acc[NE];
#pragma unroll
  for (int e = 0; e < NE; ++e) acc[e] = 0.f;
  for (int d = lane; d < DM; d += 32) {
    const float hv = row[d];
    const float* wg = w_gate + (size_t)d * NE;
#pragma unroll
    for (int e = 0; e < NE; ++e) acc[e] += hv * wg[e];
  }
#pragma unroll
  for (int e = 0; e < NE; ++e) acc[e] = wave_sum32(acc[e], lane);
  if (lane == 0) {
    float mx = acc[0];
#pragma unroll
    for (int e = 1; e < NE; ++e) mx = fmaxf(mx, acc[e]);
    float p[NE]; float Z = 0.f;
#pragma unroll
    for (int e = 0; e < NE; ++e) { p[e] = __expf(acc[e] - mx); Z += p[e]; }
    const float invZ = 1.f / Z;
#pragma unroll
    for (int e = 0; e < NE; ++e) p[e] *= invZ;
    int i0 = 0; float p0 = p[0];
#pragma unroll
    for (int e = 1; e < NE; ++e) if (p[e] > p0) { p0 = p[e]; i0 = e; }
    int i1 = 0; float p1 = -1.f;
#pragma unroll
    for (int e = 0; e < NE; ++e) if (e != i0 && p[e] > p1) { p1 = p[e]; i1 = e; }
    const float inv = 1.f / (p0 + p1 + 1e-9f);
    tidx[n] = make_int2(i0, i1);
    tgate[n] = make_float2(p0 * inv, p1 * inv);
  }
}

// ---------------- deterministic routing lists (single wave, ballot prefix) ----------------
__global__ void build_routing_kernel(const int2* __restrict__ tidx,
                                     const float2* __restrict__ tgate,
                                     int* __restrict__ counts, int* __restrict__ offsets,
                                     int* __restrict__ slot_token, float* __restrict__ slot_gate) {
  const int lane = threadIdx.x;  // 32 threads = one wave
  __shared__ int soff[NE];
  __shared__ int scnt[NE];
  int mycnt = 0;
  for (int base = 0; base < NTOK; base += 32) {
    const int2 te = tidx[base + lane];
#pragma unroll
    for (int e = 0; e < NE; ++e) {
      const unsigned m0 = __builtin_amdgcn_ballot_w32(te.x == e);
      const unsigned m1 = __builtin_amdgcn_ballot_w32(te.y == e);
      if (lane == e) mycnt += __popc(m0) + __popc(m1);
    }
  }
  if (lane < NE) { counts[lane] = mycnt; scnt[lane] = mycnt; }
  __syncthreads();
  if (lane == 0) {
    int o = 0;
    for (int e = 0; e < NE; ++e) { soff[e] = o; offsets[e] = o; o += scnt[e]; }
    offsets[NE] = o;
  }
  __syncthreads();
  const unsigned ltmask = (1u << lane) - 1u;
  for (int base = 0; base < NTOK; base += 32) {
    const int n = base + lane;
    const int2 te = tidx[n];
    const float2 tg = tgate[n];
#pragma unroll
    for (int e = 0; e < NE; ++e) {
      const unsigned m0 = __builtin_amdgcn_ballot_w32(te.x == e);
      if (te.x == e) {
        const int p = soff[e] + __popc(m0 & ltmask);
        slot_token[p] = n; slot_gate[p] = tg.x;
      }
      __syncthreads();
      if (lane == e) soff[e] += __popc(m0);
      __syncthreads();
      const unsigned m1 = __builtin_amdgcn_ballot_w32(te.y == e);
      if (te.y == e) {
        const int p = soff[e] + __popc(m1 & ltmask);
        slot_token[p] = n; slot_gate[p] = tg.y;
      }
      __syncthreads();
      if (lane == e) soff[e] += __popc(m1);
      __syncthreads();
    }
  }
}

// ---------------- grouped expert GEMM1: act = gelu(gather(h) @ w1[e] + b1[e]) ----------------
__global__ __launch_bounds__(256) void moe_gemm1_kernel(
    const __bf16* __restrict__ hb,   // [NTOK][DM]
    const __bf16* __restrict__ w1t,  // [NE][DH][DM]  (N-major)
    const float* __restrict__ b1,    // [NE][DH]
    const int* __restrict__ slot_token,
    const int* __restrict__ counts, const int* __restrict__ offsets,
    __bf16* __restrict__ act) {      // [NSLOT][DH]
  const int e = blockIdx.z;
  const int rows_e = counts[e];
  const int mtile = blockIdx.x * BM;
  if (mtile >= rows_e) return;                 // uniform: EXEC stays all-ones for WMMA
  const int off_e = offsets[e];
  const int ntile = blockIdx.y * BN;
  __shared__ __bf16 As[BM * ROWP];
  __shared__ __bf16 Bs[BN * ROWP];
  const int tid = threadIdx.x, lane = tid & 31, wid = tid >> 5;
  const int wm = wid >> 2, wn = wid & 3;
  const __bf16* Bt = w1t + (size_t)e * DH * DM;
  v8f acc[4][2] = {};
  for (int kb = 0; kb < DM; kb += KT) {
#pragma unroll
    for (int it = 0; it < 2; ++it) {
      const int ch = tid + it * 256, row = ch >> 2, cc = ch & 3;
      int r = mtile + row; if (r >= rows_e) r = rows_e - 1;   // clamp pad rows
      const int tok = slot_token[off_e + r];
      cp_b128(hb + (size_t)tok * DM + kb + cc * 8, As + row * ROWP + cc * 8);
      cp_b128(Bt + (size_t)(ntile + row) * DM + kb + cc * 8, Bs + row * ROWP + cc * 8);
    }
    stage_wait();
    v16bf af[4], bfr[2];
#pragma unroll
    for (int i = 0; i < 4; ++i) af[i] = frag_load(As, wm * 64 + i * 16, lane);
#pragma unroll
    for (int j = 0; j < 2; ++j) bfr[j] = frag_load(Bs, wn * 32 + j * 16, lane);
#pragma unroll
    for (int i = 0; i < 4; ++i)
#pragma unroll
      for (int j = 0; j < 2; ++j)
        acc[i][j] = __builtin_amdgcn_wmma_f32_16x16x32_bf16(false, af[i], false, bfr[j],
                                                            (short)0, acc[i][j], false, false);
    __syncthreads();
  }
#pragma unroll
  for (int i = 0; i < 4; ++i)
#pragma unroll
    for (int j = 0; j < 2; ++j) {
      const int mbase = mtile + wm * 64 + i * 16 + ((lane >> 4) << 3);
      const int col = ntile + wn * 32 + j * 16 + (lane & 15);
      const float b = b1[(size_t)e * DH + col];
#pragma unroll
      for (int r = 0; r < 8; ++r) {
        const int m = mbase + r;
        if (m < rows_e)
          act[(size_t)(off_e + m) * DH + col] = (__bf16)gelu_tanh(acc[i][j][r] + b);
      }
    }
}

// ---------------- grouped expert GEMM2: mixed[token] += gate * (act @ w2[e] + b2[e]) ----------------
__global__ __launch_bounds__(256) void moe_gemm2_kernel(
    const __bf16* __restrict__ act,  // [NSLOT][DH]
    const __bf16* __restrict__ w2t,  // [NE][DM][DH]  (N-major)
    const float* __restrict__ b2,    // [NE][DM]
    const int* __restrict__ slot_token, const float* __restrict__ slot_gate,
    const int* __restrict__ counts, const int* __restrict__ offsets,
    float* __restrict__ mixed) {     // [NTOK][DM]  (zero-initialized)
  const int e = blockIdx.z;
  const int rows_e = counts[e];
  const int mtile = blockIdx.x * BM;
  if (mtile >= rows_e) return;
  const int off_e = offsets[e];
  const int ntile = blockIdx.y * BN;
  __shared__ __bf16 As[BM * ROWP];
  __shared__ __bf16 Bs[BN * ROWP];
  const int tid = threadIdx.x, lane = tid & 31, wid = tid >> 5;
  const int wm = wid >> 2, wn = wid & 3;
  const __bf16* Bt = w2t + (size_t)e * DM * DH;
  v8f acc[4][2] = {};
  for (int kb = 0; kb < DH; kb += KT) {
#pragma unroll
    for (int it = 0; it < 2; ++it) {
      const int ch = tid + it * 256, row = ch >> 2, cc = ch & 3;
      int r = mtile + row; if (r >= rows_e) r = rows_e - 1;
      cp_b128(act + (size_t)(off_e + r) * DH + kb + cc * 8, As + row * ROWP + cc * 8);
      cp_b128(Bt + (size_t)(ntile + row) * DH + kb + cc * 8, Bs + row * ROWP + cc * 8);
    }
    stage_wait();
    v16bf af[4], bfr[2];
#pragma unroll
    for (int i = 0; i < 4; ++i) af[i] = frag_load(As, wm * 64 + i * 16, lane);
#pragma unroll
    for (int j = 0; j < 2; ++j) bfr[j] = frag_load(Bs, wn * 32 + j * 16, lane);
#pragma unroll
    for (int i = 0; i < 4; ++i)
#pragma unroll
      for (int j = 0; j < 2; ++j)
        acc[i][j] = __builtin_amdgcn_wmma_f32_16x16x32_bf16(false, af[i], false, bfr[j],
                                                            (short)0, acc[i][j], false, false);
    __syncthreads();
  }
#pragma unroll
  for (int i = 0; i < 4; ++i)
#pragma unroll
    for (int j = 0; j < 2; ++j) {
      const int mbase = mtile + wm * 64 + i * 16 + ((lane >> 4) << 3);
      const int col = ntile + wn * 32 + j * 16 + (lane & 15);
      const float b = b2[(size_t)e * DM + col];
#pragma unroll
      for (int r = 0; r < 8; ++r) {
        const int m = mbase + r;
        if (m < rows_e) {
          const int s = off_e + m;
          // each token receives exactly 2 commutative f32 atomic adds -> deterministic
          atomicAdd(&mixed[(size_t)slot_token[s] * DM + col], slot_gate[s] * (acc[i][j][r] + b));
        }
      }
    }
}

// ---------------- fused residual + LN + LN + 1024->2 classifier ----------------
__global__ __launch_bounds__(256) void head_kernel(
    const float* __restrict__ h, const float* __restrict__ mixed,
    const float* __restrict__ g1, const float* __restrict__ be1,
    const float* __restrict__ g2, const float* __restrict__ be2,
    const float* __restrict__ w_cls, const float* __restrict__ b_cls,
    float* __restrict__ out) {
  const int n = blockIdx.x, t = threadIdx.x;
  __shared__ float r1[256], r2[256];
  float y[4];
  float s = 0.f, ss = 0.f;
#pragma unroll
  for (int j = 0; j < 4; ++j) {
    const int d = t + 256 * j;
    const float v = h[(size_t)n * DM + d] + mixed[(size_t)n * DM + d];
    y[j] = v; s += v; ss += v * v;
  }
  float S, SS;
  block_reduce2(s, ss, r1, r2, S, SS, t);
  float mean = S * (1.f / DM);
  float rs = rsqrtf(SS * (1.f / DM) - mean * mean + 1e-5f);
  float z[4]; s = 0.f; ss = 0.f;
#pragma unroll
  for (int j = 0; j < 4; ++j) {
    const int d = t + 256 * j;
    const float v = (y[j] - mean) * rs * g1[d] + be1[d];
    z[j] = v; s += v; ss += v * v;
  }
  block_reduce2(s, ss, r1, r2, S, SS, t);
  mean = S * (1.f / DM);
  rs = rsqrtf(SS * (1.f / DM) - mean * mean + 1e-5f);
  float c0 = 0.f, c1 = 0.f;
#pragma unroll
  for (int j = 0; j < 4; ++j) {
    const int d = t + 256 * j;
    const float v = (z[j] - mean) * rs * g2[d] + be2[d];
    c0 += v * w_cls[(size_t)d * 2 + 0];
    c1 += v * w_cls[(size_t)d * 2 + 1];
  }
  block_reduce2(c0, c1, r1, r2, S, SS, t);
  if (t == 0) {
    out[(size_t)n * 2 + 0] = S  + b_cls[0];
    out[(size_t)n * 2 + 1] = SS + b_cls[1];
  }
}

// ---------------- host side ----------------
extern "C" void kernel_launch(void* const* d_in, const int* in_sizes, int n_in,
                              void* d_out, int out_size, void* d_ws, size_t ws_size,
                              hipStream_t stream) {
  (void)in_sizes; (void)n_in; (void)out_size; (void)ws_size;
  const float* x        = (const float*)d_in[0];
  const float* w_in     = (const float*)d_in[1];
  const float* b_in     = (const float*)d_in[2];
  const float* g_in     = (const float*)d_in[3];
  const float* beta_in  = (const float*)d_in[4];
  const float* w_gate   = (const float*)d_in[5];
  const float* w1       = (const float*)d_in[6];
  const float* b1       = (const float*)d_in[7];
  const float* w2       = (const float*)d_in[8];
  const float* b2       = (const float*)d_in[9];
  const float* g_moe    = (const float*)d_in[10];
  const float* beta_moe = (const float*)d_in[11];
  const float* g_out    = (const float*)d_in[12];
  const float* beta_out = (const float*)d_in[13];
  const float* w_cls    = (const float*)d_in[14];
  const float* b_cls    = (const float*)d_in[15];
  float* out = (float*)d_out;

  char* wsp = (char*)d_ws;
  auto alloc = [&](size_t bytes) -> char* {
    char* p = wsp; wsp += (bytes + 255) & ~(size_t)255; return p;
  };
  __bf16* xb         = (__bf16*)alloc((size_t)NTOK * DIN * 2);
  __bf16* w_in_t     = (__bf16*)alloc((size_t)DM * DIN * 2);
  __bf16* w1_t       = (__bf16*)alloc((size_t)NE * DH * DM * 2);
  __bf16* w2_t       = (__bf16*)alloc((size_t)NE * DM * DH * 2);
  float*  hpre       = (float*)alloc((size_t)NTOK * DM * 4);
  float*  h          = (float*)alloc((size_t)NTOK * DM * 4);
  __bf16* hb         = (__bf16*)alloc((size_t)NTOK * DM * 2);
  int2*   tidx       = (int2*)alloc((size_t)NTOK * sizeof(int2));
  float2* tgate      = (float2*)alloc((size_t)NTOK * sizeof(float2));
  int*    counts     = (int*)alloc(64);
  int*    offsets    = (int*)alloc(64);
  int*    slot_token = (int*)alloc((size_t)NSLOT * 4);
  float*  slot_gate  = (float*)alloc((size_t)NSLOT * 4);
  __bf16* act        = (__bf16*)alloc((size_t)NSLOT * DH * 2);
  float*  mixed      = (float*)alloc((size_t)NTOK * DM * 4);

  const dim3 blk(256);
  f32_to_bf16_kernel<<<(NTOK * DIN + 255) / 256, blk, 0, stream>>>(x, xb, NTOK * DIN);
  transpose_to_bf16_kernel<<<dim3(DIN / 256, DM, 1), blk, 0, stream>>>(w_in, w_in_t, DIN, DM);
  transpose_to_bf16_kernel<<<dim3(DM / 256, DH, NE), blk, 0, stream>>>(w1, w1_t, DM, DH);
  transpose_to_bf16_kernel<<<dim3(DH / 256, DM, NE), blk, 0, stream>>>(w2, w2_t, DH, DM);
  gemm_in_kernel<<<dim3(NTOK / BM, DM / BN), blk, 0, stream>>>(xb, w_in_t, b_in, hpre);
  ln_in_kernel<<<NTOK, blk, 0, stream>>>(hpre, g_in, beta_in, h, hb);
  router_kernel<<<NTOK / 8, blk, 0, stream>>>(h, w_gate, tidx, tgate);
  build_routing_kernel<<<1, 32, 0, stream>>>(tidx, tgate, counts, offsets, slot_token, slot_gate);
  zero_f32_kernel<<<2048, blk, 0, stream>>>((float4*)mixed, NTOK * DM / 4);
  moe_gemm1_kernel<<<dim3(NTOK / BM, DH / BN, NE), blk, 0, stream>>>(
      hb, w1_t, b1, slot_token, counts, offsets, act);
  moe_gemm2_kernel<<<dim3(NTOK / BM, DM / BN, NE), blk, 0, stream>>>(
      act, w2_t, b2, slot_token, slot_gate, counts, offsets, mixed);
  head_kernel<<<NTOK, blk, 0, stream>>>(h, mixed, g_moe, beta_moe, g_out, beta_out,
                                        w_cls, b_cls, out);
}
